// DeltaNet_154618823093
// MI455X (gfx1250) — matrix operand order repaired and checked by reading, not compiled, not run on hardware
//
#include <hip/hip_runtime.h>
#include <hip/hip_bf16.h>
#include <math.h>

// ---------------- types ----------------
typedef __bf16 bf16_t;
typedef __attribute__((ext_vector_type(16))) __bf16 v16bf;
typedef __attribute__((ext_vector_type(8)))  float  v8f;

// Problem constants (from reference setup_inputs)
#define BATCH 2
#define SEQ   4096
#define DIM   1024
#define NH    16
#define HD    64          // dk == dv == 64
#define NCH   64          // SEQ / 64 chunks
#define CH    64          // chunk length

// ---------------- WMMA helpers ----------------
__device__ __forceinline__ v8f wmma_bf16(v16bf a, v16bf b, v8f c) {
  // D = A(16x32) * B(32x16) + C, f32 accumulate
  return __builtin_amdgcn_wmma_f32_16x16x32_bf16(false, a, false, b, (short)0, c, false, false);
}

// A fragment: 16x32 bf16 tile at (row0,k0) of row-major matrix, ld = row stride.
// ISA layout: lanes 0-15 row M=lane hold K {0..7,16..23}; lanes 16-31 hold K {8..15,24..31}.
__device__ __forceinline__ v16bf load_a_frag(const bf16_t* p, int ld, int lane) {
  int m = lane & 15;
  int h = (lane >> 4) * 8;
  const bf16_t* r = p + m * ld + h;
  v16bf f;
#pragma unroll
  for (int i = 0; i < 8; ++i) f[i] = r[i];
#pragma unroll
  for (int i = 0; i < 8; ++i) f[8 + i] = r[16 + i];
  return f;
}

// B fragment for B = Y^T: Y row-major [N x K]; lane holds column n = lane&15,
// K {0..7,16..23} for lanes 0-15, K {8..15,24..31} for lanes 16-31 (matches A).
__device__ __forceinline__ v16bf load_bT_frag(const bf16_t* y, int ld, int lane) {
  int n  = lane & 15;
  int kb = (lane >> 4) * 8;
  const bf16_t* r = y + n * ld + kb;
  v16bf f;
#pragma unroll
  for (int i = 0; i < 8; ++i) f[i] = r[i];
#pragma unroll
  for (int i = 0; i < 8; ++i) f[8 + i] = r[16 + i];
  return f;
}

__device__ __forceinline__ float silu_f(float x) {
  return x / (1.0f + __expf(-x));
}

// ---------------- Kernel 1: fused q/k/v projection GEMM ----------------
// X [8192,1024] f32 @ W [1024,1024] f32 -> bf16 [b,h,l,64], double-buffered LDS.
#define BM 128
#define BN 128
#define BKK 32

__device__ __forceinline__ void stage_tiles(
    const float* __restrict__ X, const float* __restrict__ W,
    int mt, int nt, int kb, bf16_t* As, bf16_t* Bt, int tid)
{
#pragma unroll 2
  for (int e = tid; e < BM * BKK; e += 256) {
    int i = e >> 5, j = e & 31;
    As[i * BKK + j] = (bf16_t)X[(size_t)(mt * BM + i) * DIM + kb + j];
  }
#pragma unroll 2
  for (int e = tid; e < BKK * BN; e += 256) {
    int k = e >> 7, n = e & 127;
    Bt[n * BKK + k] = (bf16_t)W[(size_t)(kb + k) * DIM + nt * BN + n];
  }
}

__global__ __launch_bounds__(256) void proj_gemm(
    const float* __restrict__ X,
    const float* __restrict__ Wq, const float* __restrict__ Wk,
    const float* __restrict__ Wv,
    bf16_t* __restrict__ qs, bf16_t* __restrict__ ks, bf16_t* __restrict__ vs)
{
  __shared__ bf16_t As[2][BM * BKK];   // 2 x 8 KB
  __shared__ bf16_t Bt[2][BN * BKK];   // 2 x 8 KB

  const int mt = blockIdx.x, nt = blockIdx.y, mat = blockIdx.z;
  const float* W   = (mat == 0) ? Wq : (mat == 1 ? Wk : Wv);
  bf16_t*      Dst = (mat == 0) ? qs : (mat == 1 ? ks : vs);

  const int tid = threadIdx.x, lane = tid & 31, wv = tid >> 5;
  const int wm = wv >> 2;   // 0..1 : 64-row strip
  const int wn = wv & 3;    // 0..3 : 32-col strip

  v8f acc[4][2] = {};

  stage_tiles(X, W, mt, nt, 0, As[0], Bt[0], tid);
  __syncthreads();

  int p = 0;
  for (int kb = 0; kb < DIM; kb += BKK) {
    if (kb + BKK < DIM)   // overlap next-tile global loads with WMMA
      stage_tiles(X, W, mt, nt, kb + BKK, As[p ^ 1], Bt[p ^ 1], tid);

    v16bf bfrag[2];
#pragma unroll
    for (int bb = 0; bb < 2; ++bb)
      bfrag[bb] = load_bT_frag(&Bt[p][(wn * 32 + bb * 16) * BKK], BKK, lane);
#pragma unroll
    for (int a = 0; a < 4; ++a) {
      v16bf af = load_a_frag(&As[p][(wm * 64 + a * 16) * BKK], BKK, lane);
#pragma unroll
      for (int bb = 0; bb < 2; ++bb)
        acc[a][bb] = wmma_bf16(af, bfrag[bb], acc[a][bb]);
    }
    __syncthreads();
    p ^= 1;
  }

  // epilogue: silu on q/k, q *= dk^-0.5, scatter to [b,h,l,64] bf16
  const int mlo = (lane >> 4) * 8;
  const int nn  = lane & 15;
#pragma unroll
  for (int a = 0; a < 4; ++a)
#pragma unroll
    for (int bb = 0; bb < 2; ++bb)
#pragma unroll
      for (int r = 0; r < 8; ++r) {
        float v = acc[a][bb][r];
        int row = mt * BM + wm * 64 + a * 16 + r + mlo;   // b*4096 + t
        int col = nt * BN + wn * 32 + bb * 16 + nn;       // h*64 + d
        if (mat == 0)      v = silu_f(v) * 0.125f;        // q: silu, dk^-1/2
        else if (mat == 1) v = silu_f(v);                 // k: silu
        int bi = row >> 12, t = row & (SEQ - 1);
        int h  = col >> 6,  d = col & 63;
        Dst[(((size_t)(bi * NH + h) * SEQ) + t) * HD + d] = (bf16_t)v;
      }
}

// ---------------- Kernel 2: beta projection ----------------
__global__ __launch_bounds__(256) void beta_proj(
    const float* __restrict__ X, const float* __restrict__ Wb,
    float* __restrict__ beta)
{
  int idx = blockIdx.x * 256 + threadIdx.x;          // 8192*16 threads
  int r = idx >> 4, h = idx & 15;
  const float* x = X + (size_t)r * DIM;
  float acc = 0.0f;
#pragma unroll 8
  for (int k = 0; k < DIM; ++k) acc += x[k] * Wb[k * NH + h];
  int bi = r >> 12, t = r & (SEQ - 1);
  beta[((size_t)(bi * NH + h)) * SEQ + t] = acc;
}

// ---------------- Kernel 3: parallel per-chunk prep ----------------
// Grid = b*h*nchunks = 2048 blocks (fills the chip). Computes, per 64x64 chunk:
//   Asl = strict(k_b k^T); solve A [w|u] = [v_b|k_b]; attn = tril(q k^T); kT.
// Writes: w (bf16), -u (bf16), attn over the v buffer, kT over the k buffer.
__global__ __launch_bounds__(256) void chunk_prep(
    const bf16_t* __restrict__ qs,
    bf16_t* __restrict__ ks,   // in: k, out: k^T (in place, per chunk)
    bf16_t* __restrict__ vs,   // in: v, out: masked attn (in place)
    const float* __restrict__ beta,
    bf16_t* __restrict__ wbuf, bf16_t* __restrict__ ubuf /* stores -u */)
{
  extern __shared__ __align__(16) char smem[];
  bf16_t* kbf = (bf16_t*)smem;            // 8192  k chunk
  bf16_t* kbb = kbf + 4096;               // 8192  k*beta
  float*  Asl = (float*)(smem + 16384);   // 16384 strict-lower kb k^T
  float*  P   = (float*)(smem + 32768);   // 32768 [64][128] = [w | u]
  float*  bL  = (float*)(smem + 65536);   // 256   beta chunk
  // total 65792 bytes

  const int bid = blockIdx.x;
  const int bh = bid >> 6, c = bid & 63;
  const size_t cb = ((size_t)bh * SEQ + (size_t)c * CH) * HD;
  const bf16_t* qg = qs + cb;

  const int tid = threadIdx.x, lane = tid & 31, wv = tid >> 5;
  const int mlo = (lane >> 4) * 8, nn = lane & 15;

  // stage k chunk + beta
  for (int e = tid; e < 4096; e += 256) kbf[e] = ks[cb + e];
  if (tid < 64) bL[tid] = beta[(size_t)bh * SEQ + c * CH + tid];
  __syncthreads();

  // build k*beta, RHS P = [v*beta | k*beta]; write k^T back to global (in place)
  for (int e = tid; e < 4096; e += 256) {
    int i = e >> 6, d = e & 63;
    float kv = (float)kbf[e];
    float bv = bL[i];
    kbb[e]              = (bf16_t)(kv * bv);
    P[i * 128 + 64 + d] = kv * bv;
    P[i * 128 + d]      = (float)vs[cb + e] * bv;
    ks[cb + d * 64 + i] = kbf[e];          // k^T
  }
  __syncthreads();

  // Asl = (k*beta) @ k^T ; attn = tril(q @ k^T) -> overwrite v buffer
#pragma unroll
  for (int t = 0; t < 2; ++t) {
    int tile = wv * 2 + t, tm = tile >> 2, tn = tile & 3;
    v8f acc = {};
    v8f at  = {};
#pragma unroll
    for (int k0 = 0; k0 < 64; k0 += 32) {
      v16bf bk = load_bT_frag(kbf + (tn * 16) * 64 + k0, 64, lane);
      acc = wmma_bf16(load_a_frag(kbb + (tm * 16) * 64 + k0, 64, lane), bk, acc);
      at  = wmma_bf16(load_a_frag(qg  + (tm * 16) * 64 + k0, 64, lane), bk, at);
    }
#pragma unroll
    for (int r = 0; r < 8; ++r) {
      int row = tm * 16 + r + mlo, col = tn * 16 + nn;
      Asl[row * 64 + col] = acc[r];
      vs[cb + (size_t)row * 64 + col] = (bf16_t)((col <= row) ? at[r] : 0.0f);
    }
  }
  __syncthreads();

  // unit-lower-triangular solve, per-column, no inner barriers
  if (tid < 128) {
    int col = tid;
    for (int i = 0; i < 63; ++i) {
      float pi = P[i * 128 + col];
      for (int ii = i + 1; ii < 64; ++ii)
        P[ii * 128 + col] -= Asl[ii * 64 + i] * pi;
    }
  }
  __syncthreads();

  // emit w and -u (negated so the scan can use WMMA C-accumulate for w - u@S)
  for (int e = tid; e < 4096; e += 256) {
    int i = e >> 6, d = e & 63;
    wbuf[cb + e] = (bf16_t)(P[i * 128 + d]);
    ubuf[cb + e] = (bf16_t)(-P[i * 128 + 64 + d]);
  }
}

// ---------------- Kernel 4: minimal sequential scan ----------------
// One workgroup per (b,h). S lives in per-wave registers (2 tiles each);
// only S^T (bf16) goes through LDS for cross-wave B operands. 3 barriers/chunk.
__global__ __launch_bounds__(256) void delta_scan(
    const bf16_t* __restrict__ qs, const bf16_t* __restrict__ kTs,
    const bf16_t* __restrict__ attb, const bf16_t* __restrict__ wb,
    const bf16_t* __restrict__ ub, float* __restrict__ out)
{
  __shared__ bf16_t STb[4096];   // 8 KB  S^T
  __shared__ bf16_t vnT[4096];   // 8 KB  v_new^T

  const int bh  = blockIdx.x;
  const int tid = threadIdx.x, lane = tid & 31, wv = tid >> 5;
  const int mlo = (lane >> 4) * 8, nn = lane & 15;
  const size_t base0 = (size_t)bh * SEQ * HD;

  v8f Sreg[2] = {};   // this wave's two 16x16 tiles of S (f32, persistent)

  for (int e = tid; e < 4096; e += 256) STb[e] = (bf16_t)0.0f;
  __syncthreads();

  for (int c = 0; c < NCH; ++c) {
    const size_t cb = base0 + (size_t)c * CH * HD;
    const bf16_t* qg  = qs   + cb;
    const bf16_t* kTg = kTs  + cb;
    const bf16_t* ag  = attb + cb;
    const bf16_t* wg  = wb   + cb;
    const bf16_t* ug  = ub   + cb;

    if (c + 1 < NCH) {   // prefetch next chunk's operand streams into caches
      size_t nb = cb + (size_t)CH * HD + (size_t)tid * 16;
      __builtin_prefetch(qs   + nb, 0, 3);
      __builtin_prefetch(kTs  + nb, 0, 3);
      __builtin_prefetch(attb + nb, 0, 3);
      __builtin_prefetch(wb   + nb, 0, 3);
      __builtin_prefetch(ub   + nb, 0, 3);
    }

    // phase A: v_new = w + (-u) @ S  -> store v_new^T (bf16)
#pragma unroll
    for (int t = 0; t < 2; ++t) {
      int tile = wv * 2 + t, tm = tile >> 2, tn = tile & 3;
      v8f acc;
#pragma unroll
      for (int r = 0; r < 8; ++r)
        acc[r] = (float)wg[(size_t)(tm * 16 + r + mlo) * 64 + tn * 16 + nn];
#pragma unroll
      for (int k0 = 0; k0 < 64; k0 += 32)
        acc = wmma_bf16(load_a_frag(ug + (tm * 16) * 64 + k0, 64, lane),
                        load_bT_frag(STb + (tn * 16) * 64 + k0, 64, lane), acc);
#pragma unroll
      for (int r = 0; r < 8; ++r)
        vnT[(tn * 16 + nn) * 64 + tm * 16 + r + mlo] = (bf16_t)acc[r];
    }
    __syncthreads();

    // phase B: o = q@S + attn@v_new (write out); S += k^T @ v_new (registers)
#pragma unroll
    for (int t = 0; t < 2; ++t) {
      int tile = wv * 2 + t, tm = tile >> 2, tn = tile & 3;
      v8f oacc = {};
#pragma unroll
      for (int k0 = 0; k0 < 64; k0 += 32)
        oacc = wmma_bf16(load_a_frag(qg + (tm * 16) * 64 + k0, 64, lane),
                         load_bT_frag(STb + (tn * 16) * 64 + k0, 64, lane), oacc);
#pragma unroll
      for (int k0 = 0; k0 < 64; k0 += 32)
        oacc = wmma_bf16(load_a_frag(ag + (tm * 16) * 64 + k0, 64, lane),
                         load_bT_frag(vnT + (tn * 16) * 64 + k0, 64, lane), oacc);
#pragma unroll
      for (int r = 0; r < 8; ++r)
        out[cb + (size_t)(tm * 16 + r + mlo) * 64 + tn * 16 + nn] = oacc[r];

#pragma unroll
      for (int k0 = 0; k0 < 64; k0 += 32)
        Sreg[t] = wmma_bf16(load_a_frag(kTg + (tm * 16) * 64 + k0, 64, lane),
                            load_bT_frag(vnT + (tn * 16) * 64 + k0, 64, lane),
                            Sreg[t]);
    }
    __syncthreads();   // all reads of old STb complete

    // phase C: publish S^T for next chunk
#pragma unroll
    for (int t = 0; t < 2; ++t) {
      int tile = wv * 2 + t, tm = tile >> 2, tn = tile & 3;
#pragma unroll
      for (int r = 0; r < 8; ++r)
        STb[(tn * 16 + nn) * 64 + tm * 16 + r + mlo] = (bf16_t)Sreg[t][r];
    }
    __syncthreads();
  }
}

// ---------------- launch ----------------
extern "C" void kernel_launch(void* const* d_in, const int* in_sizes, int n_in,
                              void* d_out, int out_size, void* d_ws, size_t ws_size,
                              hipStream_t stream) {
  (void)in_sizes; (void)n_in; (void)out_size; (void)ws_size;
  const float* X  = (const float*)d_in[0];
  const float* Wq = (const float*)d_in[1];
  const float* Wk = (const float*)d_in[2];
  const float* Wv = (const float*)d_in[3];
  const float* Wb = (const float*)d_in[4];
  float* out = (float*)d_out;

  const size_t nQ = (size_t)BATCH * NH * SEQ * HD;   // 8388608 elements
  bf16_t* qs   = (bf16_t*)d_ws;
  bf16_t* ksb  = qs  + nQ;          // k, becomes k^T
  bf16_t* vsb  = ksb + nQ;          // v, becomes attn
  bf16_t* wbuf = vsb + nQ;          // w
  bf16_t* ubuf = wbuf + nQ;         // -u
  float*  betb = (float*)(ubuf + nQ);

  proj_gemm<<<dim3(SEQ * BATCH / BM, DIM / BN, 3), 256, 0, stream>>>(
      X, Wq, Wk, Wv, qs, ksb, vsb);
  beta_proj<<<(BATCH * SEQ * NH) / 256, 256, 0, stream>>>(X, Wb, betb);
  chunk_prep<<<BATCH * NH * NCH, 256, 65792, stream>>>(
      qs, ksb, vsb, betb, wbuf, ubuf);
  delta_scan<<<BATCH * NH, 256, 0, stream>>>(qs, ksb, vsb, wbuf, ubuf, out);
}
